// MultiheadAttention_50981261804083
// MI455X (gfx1250) — matrix operand order repaired and checked by reading, not compiled, not run on hardware
//
#include <hip/hip_runtime.h>
#include <hip/hip_bf16.h>

#define DMODEL 1024
#define SEQ    2048
#define NH     16
#define DKH    64
#define NB     2
#define MTOT   (NB * SEQ)   // 4096 token rows
#define BHN    (NB * NH)    // 32 batch*heads

typedef __bf16 bf16_t;
typedef __attribute__((ext_vector_type(16))) __bf16 v16bf;
typedef __attribute__((ext_vector_type(8)))  float  v8f;

union BFrag {
  v16bf bf;
  uint4 q[2];
};

// ---------------------------------------------------------------------------
// Fragment loaders. WMMA bf16 operand layouts (ISA 7.12.2):
//   A (16x32): lane (m=lane&15, hf=lane>>4) holds K in [kk+8hf, +8) U [kk+16+8hf, +8)
//   B (32x16): lane (n=lane&15, hf=lane>>4) holds K in [kk+8hf, +8) U [kk+16+8hf, +8), col n
// ---------------------------------------------------------------------------
__device__ __forceinline__ BFrag cvt16(const float* t) {
  BFrag f;
#pragma unroll
  for (int i = 0; i < 16; ++i) f.bf[i] = (__bf16)t[i];  // v_cvt_pk_bf16_f32
  return f;
}

__device__ __forceinline__ BFrag load_a_f32(const float* __restrict__ base, int ld,
                                            int row, int kk, int hf) {
  const float* rp = base + (size_t)row * ld + kk + hf * 8;
  float t[16];
  *(float4*)(t + 0)  = *(const float4*)(rp + 0);
  *(float4*)(t + 4)  = *(const float4*)(rp + 4);
  *(float4*)(t + 8)  = *(const float4*)(rp + 16);
  *(float4*)(t + 12) = *(const float4*)(rp + 20);
  return cvt16(t);
}

__device__ __forceinline__ BFrag load_a_bf16(const bf16_t* __restrict__ base, int ld,
                                             int row, int kk, int hf) {
  const bf16_t* rp = base + (size_t)row * ld + kk + hf * 8;
  BFrag f;
  f.q[0] = *(const uint4*)(rp + 0);
  f.q[1] = *(const uint4*)(rp + 16);
  return f;
}

__device__ __forceinline__ BFrag load_b_bf16(const bf16_t* __restrict__ base, int ld,
                                             int col, int kk, int hf) {
  const bf16_t* rp = base + (size_t)col * ld + kk + hf * 8;
  BFrag f;
  f.q[0] = *(const uint4*)(rp + 0);
  f.q[1] = *(const uint4*)(rp + 16);
  return f;
}

// B-fragment read from an LDS-staged [col][64 K] bf16 tile (row stride 64).
__device__ __forceinline__ BFrag load_b_lds64(const bf16_t* tile, int ncol, int ks, int hf) {
  const bf16_t* rp = tile + ncol * 64 + ks + hf * 8;
  BFrag f;
  f.q[0] = *(const uint4*)(rp + 0);
  f.q[1] = *(const uint4*)(rp + 16);
  return f;
}

// Cooperative stage of a 64x64 fp32 tile (64 output cols x 64 K) into LDS bf16.
// 256 threads: thread t loads 16 floats (4x float4), converts, stores 2x uint4.
__device__ __forceinline__ void stage64x64_f32(const float* __restrict__ W, int ld,
                                               int colBase, int kk, bf16_t* dst, int tid) {
  int c = tid >> 2;         // 0..63
  int kq = (tid & 3) * 16;  // 0,16,32,48
  const float* rp = W + (size_t)(colBase + c) * ld + kk + kq;
  float t[16];
  *(float4*)(t + 0)  = *(const float4*)(rp + 0);
  *(float4*)(t + 4)  = *(const float4*)(rp + 4);
  *(float4*)(t + 8)  = *(const float4*)(rp + 8);
  *(float4*)(t + 12) = *(const float4*)(rp + 12);
  union { __bf16 b[16]; uint4 u[2]; } p;
#pragma unroll
  for (int i = 0; i < 16; ++i) p.b[i] = (__bf16)t[i];
  *(uint4*)(dst + c * 64 + kq + 0) = p.u[0];
  *(uint4*)(dst + c * 64 + kq + 8) = p.u[1];
}

// Same but source already bf16 (pure 32B copy per thread).
__device__ __forceinline__ void stage64x64_bf16(const bf16_t* __restrict__ W, int ld,
                                                int colBase, int kk, bf16_t* dst, int tid) {
  int c = tid >> 2;
  int kq = (tid & 3) * 16;
  const bf16_t* rp = W + (size_t)(colBase + c) * ld + kk + kq;
  *(uint4*)(dst + c * 64 + kq + 0) = *(const uint4*)(rp + 0);
  *(uint4*)(dst + c * 64 + kq + 8) = *(const uint4*)(rp + 8);
}

__device__ __forceinline__ v8f wmma_bf16(BFrag a, BFrag b, v8f c) {
  return __builtin_amdgcn_wmma_f32_16x16x32_bf16(false, a.bf, false, b.bf,
                                                 (short)0, c, false, false);
}

// ---------------------------------------------------------------------------
// K1: QKV projections, 32x64 per wave (2x4 tiles). 64x64 weight tile staged in
// LDS (shared by all 8 waves), double-buffered; 16 WMMAs per barrier.
// Q,K stored bf16 [b,h,s,dk]; V stored transposed bf16 [b,h,dk,s].
// ---------------------------------------------------------------------------
__global__ __launch_bounds__(256, 1) void qkv_proj_kernel(
    const float* __restrict__ xq, const float* __restrict__ xk, const float* __restrict__ xv,
    const float* __restrict__ wq, const float* __restrict__ wk, const float* __restrict__ wv,
    const float* __restrict__ bq, const float* __restrict__ bk, const float* __restrict__ bv,
    bf16_t* __restrict__ Qb, bf16_t* __restrict__ Kb, bf16_t* __restrict__ Vt) {
  __shared__ bf16_t wt[2][64 * 64];  // 16 KB double buffer

  int tid = threadIdx.x;
  int wave = tid >> 5, lane = tid & 31;
  int hf = lane >> 4, l16 = lane & 15;
  int mt2 = blockIdx.x * 8 + wave;  // 0..127 (32-row macro tiles)
  int nt4 = blockIdx.y;             // 0..15  (64-col macro tiles)
  int mode = blockIdx.z;            // 0=Q 1=K 2=V

  const float* X  = (mode == 0) ? xq : (mode == 1) ? xk : xv;
  const float* W  = (mode == 0) ? wq : (mode == 1) ? wk : wv;
  const float* Bi = (mode == 0) ? bq : (mode == 1) ? bk : bv;

  int r0 = mt2 * 32 + l16, r1 = r0 + 16;
  v8f acc[2][4] = {};

  stage64x64_f32(W, DMODEL, nt4 * 64, 0, wt[0], tid);
  int buf = 0;
  for (int kk = 0; kk < DMODEL; kk += 64) {
    __syncthreads();
    if (kk + 64 < DMODEL) stage64x64_f32(W, DMODEL, nt4 * 64, kk + 64, wt[buf ^ 1], tid);
    if (kk + 128 < DMODEL) {
      __builtin_prefetch(X + (size_t)r0 * DMODEL + kk + 128, 0, 0);
      __builtin_prefetch(X + (size_t)r1 * DMODEL + kk + 128, 0, 0);
    }
#pragma unroll
    for (int ks = 0; ks < 64; ks += 32) {
      BFrag a0 = load_a_f32(X, DMODEL, r0, kk + ks, hf);
      BFrag a1 = load_a_f32(X, DMODEL, r1, kk + ks, hf);
      BFrag b[4];
#pragma unroll
      for (int tj = 0; tj < 4; ++tj) b[tj] = load_b_lds64(wt[buf], tj * 16 + l16, ks, hf);
#pragma unroll
      for (int tj = 0; tj < 4; ++tj) {
        acc[0][tj] = wmma_bf16(a0, b[tj], acc[0][tj]);
        acc[1][tj] = wmma_bf16(a1, b[tj], acc[1][tj]);
      }
    }
    buf ^= 1;
  }

#pragma unroll
  for (int ti = 0; ti < 2; ++ti) {
#pragma unroll
    for (int tj = 0; tj < 4; ++tj) {
      int n = nt4 * 64 + tj * 16 + l16;
      float bias = Bi[n];
      int h = n >> 6, dk = n & (DKH - 1);
#pragma unroll
      for (int r = 0; r < 8; ++r) {
        int m = mt2 * 32 + ti * 16 + r + hf * 8;
        int bb = m >> 11, s = m & (SEQ - 1);
        float v = acc[ti][tj][r] + bias;
        if (mode == 2) {
          Vt[(((size_t)bb * NH + h) * DKH + dk) * SEQ + s] = (bf16_t)v;
        } else {
          bf16_t* dst = (mode == 0) ? Qb : Kb;
          dst[(((size_t)bb * NH + h) * SEQ + s) * DKH + dk] = (bf16_t)v;
        }
      }
    }
  }
}

// ---------------------------------------------------------------------------
// K2: scores = Q K^T / 8, mask, softmax. One block per (bh, 16-q block).
// A-fragments reused across all 128 key tiles; scores staged bf16 in 64KB LDS.
// ---------------------------------------------------------------------------
__global__ __launch_bounds__(256) void attn_scores_kernel(
    const bf16_t* __restrict__ Qb, const bf16_t* __restrict__ Kb,
    const int* __restrict__ mask, float* __restrict__ attn) {
  __shared__ bf16_t sc[16 * SEQ];  // 64 KB

  int wave = threadIdx.x >> 5, lane = threadIdx.x & 31;
  int hf = lane >> 4, l16 = lane & 15;
  int qt = blockIdx.x;  // 0..127
  int bh = blockIdx.y;  // 0..31

  const bf16_t* Qbase = Qb + (size_t)bh * SEQ * DKH;
  const bf16_t* Kbase = Kb + (size_t)bh * SEQ * DKH;

  int qrow = qt * 16 + l16;
  BFrag a0 = load_a_bf16(Qbase, DKH, qrow, 0, hf);
  BFrag a1 = load_a_bf16(Qbase, DKH, qrow, 32, hf);

  for (int ct = wave; ct < SEQ / 16; ct += 8) {
    int key = ct * 16 + l16;
    BFrag b0 = load_b_bf16(Kbase, DKH, key, 0, hf);
    BFrag b1 = load_b_bf16(Kbase, DKH, key, 32, hf);
    v8f acc = {};
    acc = wmma_bf16(a0, b0, acc);
    acc = wmma_bf16(a1, b1, acc);
#pragma unroll
    for (int r = 0; r < 8; ++r) {
      int m = r + hf * 8;
      sc[m * SEQ + ct * 16 + l16] = (bf16_t)(acc[r] * 0.125f);  // 1/sqrt(64)
    }
  }
  __syncthreads();

  int b = bh >> 4;
  const int* mrow = mask + (size_t)b * SEQ;
  for (int rr = wave; rr < 16; rr += 8) {
    float mx = -3.0e38f;
    for (int k = lane; k < SEQ; k += 32) {
      float sv = (float)sc[rr * SEQ + k];
      sv = (mrow[k] == 0) ? -1e9f : sv;
      mx = fmaxf(mx, sv);
    }
#pragma unroll
    for (int off = 16; off > 0; off >>= 1) mx = fmaxf(mx, __shfl_xor(mx, off, 32));

    float sum = 0.f;
    for (int k = lane; k < SEQ; k += 32) {
      float sv = (float)sc[rr * SEQ + k];
      sum += (mrow[k] == 0) ? 0.f : __expf(sv - mx);
    }
#pragma unroll
    for (int off = 16; off > 0; off >>= 1) sum += __shfl_xor(sum, off, 32);

    float inv = 1.f / (sum + 1e-30f);
    float* arow = attn + ((size_t)bh * SEQ + qt * 16 + rr) * SEQ;
    for (int k = lane; k < SEQ; k += 32) {
      float sv = (float)sc[rr * SEQ + k];
      arow[k] = (mrow[k] == 0) ? 0.f : __expf(sv - mx) * inv;
    }
  }
}

// ---------------------------------------------------------------------------
// K3: context = attn @ V, 32x64 per wave (full DK in one tile). attn read fp32
// from d_out; 64x64 V tile (bf16) staged in LDS, double-buffered.
// ---------------------------------------------------------------------------
__global__ __launch_bounds__(256, 1) void context_kernel(
    const float* __restrict__ attn, const bf16_t* __restrict__ Vt,
    bf16_t* __restrict__ Ctx) {
  __shared__ bf16_t vt[2][64 * 64];  // 16 KB double buffer

  int tid = threadIdx.x;
  int wave = tid >> 5, lane = tid & 31;
  int hf = lane >> 4, l16 = lane & 15;
  int qt2 = blockIdx.x * 8 + wave;  // 0..63 (32-row macro tiles)
  int bh = blockIdx.z;              // 0..31

  const float* Ab = attn + (size_t)bh * SEQ * SEQ;
  const bf16_t* Vb = Vt + (size_t)bh * DKH * SEQ;

  int r0 = qt2 * 32 + l16, r1 = r0 + 16;
  v8f acc[2][4] = {};

  stage64x64_bf16(Vb, SEQ, 0, 0, vt[0], tid);
  int buf = 0;
  for (int kk = 0; kk < SEQ; kk += 64) {
    __syncthreads();
    if (kk + 64 < SEQ) stage64x64_bf16(Vb, SEQ, 0, kk + 64, vt[buf ^ 1], tid);
    if (kk + 128 < SEQ) {
      __builtin_prefetch(Ab + (size_t)r0 * SEQ + kk + 128, 0, 0);
      __builtin_prefetch(Ab + (size_t)r1 * SEQ + kk + 128, 0, 0);
    }
#pragma unroll
    for (int ks = 0; ks < 64; ks += 32) {
      BFrag a0 = load_a_f32(Ab, SEQ, r0, kk + ks, hf);
      BFrag a1 = load_a_f32(Ab, SEQ, r1, kk + ks, hf);
      BFrag b[4];
#pragma unroll
      for (int tj = 0; tj < 4; ++tj) b[tj] = load_b_lds64(vt[buf], tj * 16 + l16, ks, hf);
#pragma unroll
      for (int tj = 0; tj < 4; ++tj) {
        acc[0][tj] = wmma_bf16(a0, b[tj], acc[0][tj]);
        acc[1][tj] = wmma_bf16(a1, b[tj], acc[1][tj]);
      }
    }
    buf ^= 1;
  }

  int bb = bh >> 4, h = bh & 15;
#pragma unroll
  for (int ti = 0; ti < 2; ++ti) {
#pragma unroll
    for (int tj = 0; tj < 4; ++tj) {
      int n = h * DKH + tj * 16 + l16;
#pragma unroll
      for (int r = 0; r < 8; ++r) {
        int s = qt2 * 32 + ti * 16 + r + hf * 8;
        Ctx[((size_t)bb * SEQ + s) * DMODEL + n] = (bf16_t)acc[ti][tj][r];
      }
    }
  }
}

// ---------------------------------------------------------------------------
// K4: out = ctx @ Wo^T + bo + residual(query), 32x64 per wave, LDS-staged W.
// ---------------------------------------------------------------------------
__global__ __launch_bounds__(256, 1) void out_proj_kernel(
    const bf16_t* __restrict__ Ctx, const float* __restrict__ wo,
    const float* __restrict__ bo, const float* __restrict__ query,
    float* __restrict__ xres) {
  __shared__ bf16_t wt[2][64 * 64];  // 16 KB double buffer

  int tid = threadIdx.x;
  int wave = tid >> 5, lane = tid & 31;
  int hf = lane >> 4, l16 = lane & 15;
  int mt2 = blockIdx.x * 8 + wave;  // 0..127
  int nt4 = blockIdx.y;             // 0..15

  int r0 = mt2 * 32 + l16, r1 = r0 + 16;
  v8f acc[2][4] = {};

  stage64x64_f32(wo, DMODEL, nt4 * 64, 0, wt[0], tid);
  int buf = 0;
  for (int kk = 0; kk < DMODEL; kk += 64) {
    __syncthreads();
    if (kk + 64 < DMODEL) stage64x64_f32(wo, DMODEL, nt4 * 64, kk + 64, wt[buf ^ 1], tid);
    if (kk + 128 < DMODEL) {
      __builtin_prefetch(Ctx + (size_t)r0 * DMODEL + kk + 128, 0, 0);
      __builtin_prefetch(Ctx + (size_t)r1 * DMODEL + kk + 128, 0, 0);
    }
#pragma unroll
    for (int ks = 0; ks < 64; ks += 32) {
      BFrag a0 = load_a_bf16(Ctx, DMODEL, r0, kk + ks, hf);
      BFrag a1 = load_a_bf16(Ctx, DMODEL, r1, kk + ks, hf);
      BFrag b[4];
#pragma unroll
      for (int tj = 0; tj < 4; ++tj) b[tj] = load_b_lds64(wt[buf], tj * 16 + l16, ks, hf);
#pragma unroll
      for (int tj = 0; tj < 4; ++tj) {
        acc[0][tj] = wmma_bf16(a0, b[tj], acc[0][tj]);
        acc[1][tj] = wmma_bf16(a1, b[tj], acc[1][tj]);
      }
    }
    buf ^= 1;
  }

#pragma unroll
  for (int ti = 0; ti < 2; ++ti) {
#pragma unroll
    for (int tj = 0; tj < 4; ++tj) {
      int n = nt4 * 64 + tj * 16 + l16;
      float bias = bo[n];
#pragma unroll
      for (int r = 0; r < 8; ++r) {
        int m = mt2 * 32 + ti * 16 + r + hf * 8;
        size_t idx = (size_t)m * DMODEL + n;
        xres[idx] = acc[ti][tj][r] + bias + query[idx];
      }
    }
  }
}

// ---------------------------------------------------------------------------
// K5: row-wise LayerNorm -> final output
// ---------------------------------------------------------------------------
__global__ __launch_bounds__(256) void ln_kernel(
    const float* __restrict__ x, const float* __restrict__ g,
    const float* __restrict__ bt, float* __restrict__ out) {
  int row = blockIdx.x;
  const float* xr = x + (size_t)row * DMODEL;
  float s = 0.f, s2 = 0.f;
  for (int i = threadIdx.x; i < DMODEL; i += 256) {
    float v = xr[i];
    s += v;
    s2 += v * v;
  }
#pragma unroll
  for (int off = 16; off > 0; off >>= 1) {
    s += __shfl_xor(s, off, 32);
    s2 += __shfl_xor(s2, off, 32);
  }
  __shared__ float as[8], bs[8];
  int wv = threadIdx.x >> 5;
  if ((threadIdx.x & 31) == 0) {
    as[wv] = s;
    bs[wv] = s2;
  }
  __syncthreads();
  s = 0.f;
  s2 = 0.f;
#pragma unroll
  for (int i = 0; i < 8; ++i) {
    s += as[i];
    s2 += bs[i];
  }
  float mu = s * (1.0f / DMODEL);
  float var = s2 * (1.0f / DMODEL) - mu * mu;
  float rstd = rsqrtf(var + 1e-5f);
  for (int i = threadIdx.x; i < DMODEL; i += 256) {
    out[(size_t)row * DMODEL + i] = (xr[i] - mu) * rstd * g[i] + bt[i];
  }
}

// ---------------------------------------------------------------------------
extern "C" void kernel_launch(void* const* d_in, const int* in_sizes, int n_in,
                              void* d_out, int out_size, void* d_ws, size_t ws_size,
                              hipStream_t stream) {
  (void)in_sizes; (void)n_in; (void)out_size; (void)ws_size;
  const float* query = (const float*)d_in[0];
  const float* key   = (const float*)d_in[1];
  const float* value = (const float*)d_in[2];
  const int*   mask  = (const int*)d_in[3];
  const float* w_q = (const float*)d_in[4];
  const float* b_q = (const float*)d_in[5];
  const float* w_k = (const float*)d_in[6];
  const float* b_k = (const float*)d_in[7];
  const float* w_v = (const float*)d_in[8];
  const float* b_v = (const float*)d_in[9];
  const float* w_o = (const float*)d_in[10];
  const float* b_o = (const float*)d_in[11];
  const float* ln_g = (const float*)d_in[12];
  const float* ln_b = (const float*)d_in[13];

  float* out  = (float*)d_out;
  float* attn = out + (size_t)MTOT * DMODEL;  // (B,H,S,S) fp32

  const size_t NE = (size_t)MTOT * DMODEL;  // 4096*1024 elements
  bf16_t* Qb  = (bf16_t*)d_ws;   // [b,h,s,dk]
  bf16_t* Kb  = Qb + NE;         // [b,h,s,dk]
  bf16_t* Vt  = Kb + NE;         // [b,h,dk,s] (transposed)
  bf16_t* Ctx = Vt + NE;         // [B*S, D]
  float* XRes = (float*)(Ctx + NE);  // [B*S, D] fp32

  qkv_proj_kernel<<<dim3(16, 16, 3), 256, 0, stream>>>(
      query, key, value, w_q, w_k, w_v, b_q, b_k, b_v, Qb, Kb, Vt);
  attn_scores_kernel<<<dim3(128, 32), 256, 0, stream>>>(Qb, Kb, mask, attn);
  context_kernel<<<dim3(8, 1, 32), 256, 0, stream>>>(attn, Vt, Ctx);
  out_proj_kernel<<<dim3(16, 16), 256, 0, stream>>>(Ctx, w_o, b_o, query, XRes);
  ln_kernel<<<dim3(MTOT), 256, 0, stream>>>(XRes, ln_g, ln_b, out);
}
